// TTGNN_69741678952708
// MI455X (gfx1250) — matrix operand order repaired and checked by reading, not compiled, not run on hardware
//
#include <hip/hip_runtime.h>
#include <stdint.h>

// ---------------- constants (match reference) ----------------
#define NN    8192
#define EE    32768
#define DIN   1024
#define DH    256
#define NH    8
#define NL    3
#define HD    (NH*DH)     // 2048
#define E2    (EE+NN)     // 40960
#define NEG_SLOPE 0.2f

typedef unsigned short u16;
typedef unsigned int   u32;
typedef unsigned long long u64;

typedef __attribute__((ext_vector_type(8)))  float  v8f;
typedef __attribute__((ext_vector_type(16))) __bf16 v16bf;

struct B32x8 { u32 d[8]; };

// ---------------- TDM support (probe-verified builtin; guarded) ----------------
#if defined(__HIP_DEVICE_COMPILE__) && __has_builtin(__builtin_amdgcn_tensor_load_to_lds) && \
    __has_builtin(__builtin_amdgcn_s_wait_tensorcnt)
#define USE_TDM 1
#else
#define USE_TDM 0
#endif

#if USE_TDM
typedef unsigned int u32x4 __attribute__((ext_vector_type(4)));
typedef int          i32x4 __attribute__((ext_vector_type(4)));
typedef int          i32x8 __attribute__((ext_vector_type(8)));
struct U4s { u32 d[4]; };
struct I4s { int d[4]; };
struct I8s { int d[8]; };

// 2-D TDM load of a (tile_y rows x tile_x elems) bf16 tile into LDS.
// D# layout per CDNA5 ISA ch.8: group0 = {count, lds_addr, global_addr, type=2},
// group1 = {data_size=1(2B), tensor_dim0/1, tile_dim0/1, tensor_dim0_stride}.
__device__ __forceinline__ void tdm_load_2d_bf16(u32 lds_off, const void* gptr,
                                                 u32 tile_x, u32 tile_y,
                                                 u32 row_stride, u32 tensor_x, u32 tensor_y) {
    const u64 ga = (u64)gptr;
    U4s g0s;
    g0s.d[0] = 0x1u;                          // count=1 user descriptor
    g0s.d[1] = lds_off;                       // lds_addr (bytes)
    g0s.d[2] = (u32)ga;                       // global_addr[31:0]
    g0s.d[3] = (u32)(ga >> 32) | (2u << 30);  // global_addr[56:32] | type=2("image")
    I8s g1s;
    g1s.d[0] = (int)(1u << 16);               // wg_mask=0, data_size=1 (2 bytes)
    g1s.d[1] = (int)((tensor_x & 0xFFFFu) << 16);                       // tensor_dim0[15:0]
    g1s.d[2] = (int)(((tensor_x >> 16) & 0xFFFFu) | ((tensor_y & 0xFFFFu) << 16));
    g1s.d[3] = (int)(((tensor_y >> 16) & 0xFFFFu) | ((tile_x & 0xFFFFu) << 16)); // tile_dim0
    g1s.d[4] = (int)(tile_y & 0xFFFFu);       // tile_dim1 (tile_dim2=0)
    g1s.d[5] = (int)row_stride;               // tensor_dim0_stride[31:0]
    g1s.d[6] = 0;                             // stride[47:32] | tensor_dim1_stride[15:0]
    g1s.d[7] = 0;
    I4s zs; zs.d[0] = zs.d[1] = zs.d[2] = zs.d[3] = 0;
    u32x4 g0 = __builtin_bit_cast(u32x4, g0s);
    i32x8 g1 = __builtin_bit_cast(i32x8, g1s);
    i32x4 g2 = __builtin_bit_cast(i32x4, zs);
    i32x4 g3 = __builtin_bit_cast(i32x4, zs);
#if defined(__clang_major__) && (__clang_major__ >= 23)
    i32x8 g4 = (i32x8)0;
    __builtin_amdgcn_tensor_load_to_lds(g0, g1, g2, g3, g4, 0);
#else
    __builtin_amdgcn_tensor_load_to_lds(g0, g1, g2, g3, 0);
#endif
}
#endif // USE_TDM

// ---------------- helpers ----------------
__device__ __forceinline__ u16 f32_to_bf16(float f) {
    u32 u = __float_as_uint(f);
    u32 r = 0x7FFFu + ((u >> 16) & 1u);     // round to nearest even
    return (u16)((u + r) >> 16);
}
__device__ __forceinline__ u32 enc_f32(float f) {       // order-preserving uint encoding
    u32 u = __float_as_uint(f);
    return (u & 0x80000000u) ? ~u : (u | 0x80000000u);
}
__device__ __forceinline__ float dec_f32(u32 u) {
    u = (u & 0x80000000u) ? (u & 0x7FFFFFFFu) : ~u;
    return __uint_as_float(u);
}

// ---------------- elementwise kernels ----------------
__global__ void k_cvt_bf16(const float* __restrict__ in, u16* __restrict__ out, int n) {
    int i = blockIdx.x * blockDim.x + threadIdx.x;
    if (i < n) out[i] = f32_to_bf16(in[i]);
}
__global__ void k_fill_f32(float* __restrict__ p, float v, int n) {
    int i = blockIdx.x * blockDim.x + threadIdx.x;
    if (i < n) p[i] = v;
}

// per-node incoming-edge type histogram + degree (for self-loop edge_attr mean)
__global__ void k_typecnt(const int* __restrict__ dst, const int* __restrict__ attr,
                          float* __restrict__ typecnt, float* __restrict__ cnt) {
    int e = blockIdx.x * blockDim.x + threadIdx.x;
    if (e < EE) {
        int d = dst[e];
        atomicAdd(&typecnt[d * 5 + attr[e]], 1.0f);
        atomicAdd(&cnt[d], 1.0f);
    }
}

// ee_type[t][c] = sum_k edge_emb[t][k] * We_l[k][c]   (5 x 2048, tiny)
__global__ void k_edge_type_proj(const float* __restrict__ edge_emb,
                                 const float* __restrict__ We_l,
                                 float* __restrict__ ee_type) {
    int i = blockIdx.x * blockDim.x + threadIdx.x;
    if (i >= 5 * HD) return;
    int t = i / HD, c = i % HD;
    float s = 0.f;
    #pragma unroll 8
    for (int k = 0; k < DH; ++k) s = fmaf(edge_emb[t * DH + k], We_l[(size_t)k * HD + c], s);
    ee_type[i] = s;
}

// ---------------- WMMA GEMM:  C[M,Nn] = A[M,K](bf16) @ B[K,Nn](bf16) + bias ----------------
// 256 threads = 8 waves. Block tile 128x64; wave (r=wid&3, c=wid>>2) owns a 32x32 output
// (2 A-frags x 2 B-frags -> 4 WMMAs per 32-K step). A tiles stream through LDS via the
// Tensor Data Mover (double-buffered, issued by wave 0, s_wait_tensorcnt pipelined);
// B tiles are staged transposed (col-major-K) so fragment loads are contiguous ds_load_b128.
// EPI==1 adds node_emb[node_types[row]][col]  (only used when Nn==DH).
template <int EPI>
__global__ __launch_bounds__(256)
void k_wmma_gemm(const u16* __restrict__ A, const u16* __restrict__ B,
                 const float* __restrict__ bias, float* __restrict__ C,
                 int M, int K, int Nn,
                 const int* __restrict__ node_types, const float* __restrict__ node_emb) {
    __shared__ __align__(16) u16 lA[2][128 * 32];   // [buf][row][k]
    __shared__ __align__(16) u16 lB[2][64 * 32];    // [buf][col][k] (transposed)

    const int tid  = threadIdx.x;
    const int wid  = tid >> 5;
    const int lane = tid & 31;
    const int bm   = blockIdx.y * 128;
    const int bn   = blockIdx.x * 64;
    const int wm   = (wid & 3) * 32;
    const int wn   = (wid >> 2) * 32;
    const int half = lane >> 4;       // 0: lanes 0-15, 1: lanes 16-31
    const int mn   = lane & 15;

    v8f c00 = {0.f,0.f,0.f,0.f,0.f,0.f,0.f,0.f};
    v8f c01 = c00, c10 = c00, c11 = c00;

    const int nsteps = K >> 5;

    auto stageB = [&](int k0, int buf) {
        const int n = tid >> 2, kg = tid & 3;
        u16 t[8];
        #pragma unroll
        for (int i = 0; i < 8; ++i)
            t[i] = B[(size_t)(k0 + kg * 8 + i) * Nn + bn + n];
        uint4 bv;
        bv.x = (u32)t[0] | ((u32)t[1] << 16);
        bv.y = (u32)t[2] | ((u32)t[3] << 16);
        bv.z = (u32)t[4] | ((u32)t[5] << 16);
        bv.w = (u32)t[6] | ((u32)t[7] << 16);
        *(uint4*)(lB[buf] + n * 32 + kg * 8) = bv;
        if (k0 + 32 < K)   // L2 prefetch of next B K-rows (global_prefetch_b8)
            __builtin_prefetch(B + (size_t)(k0 + 32 + kg * 8) * Nn + bn + n, 0, 1);
    };

#if USE_TDM
    auto stageA = [&](int k0, int buf) {
        if (wid == 0)
            tdm_load_2d_bf16((u32)(size_t)&lA[buf][0],
                             A + (size_t)bm * K + k0,
                             /*tile_x=*/32, /*tile_y=*/128,
                             /*row_stride=*/(u32)K,
                             /*tensor_x=*/(u32)K, /*tensor_y=*/(u32)M);
    };
#else
    auto stageA = [&](int k0, int buf) {
        #pragma unroll
        for (int t2 = 0; t2 < 2; ++t2) {
            const int idx = tid + t2 * 256;
            const int r = idx >> 2, cg = idx & 3;
            const uint4 av = *(const uint4*)(A + (size_t)(bm + r) * K + k0 + cg * 8);
            *(uint4*)(lA[buf] + r * 32 + cg * 8) = av;
        }
    };
#endif

    stageA(0, 0);
    stageB(0, 0);

    for (int i = 0; i < nsteps; ++i) {
        const int cur = i & 1;
        if (i + 1 < nsteps) {
            stageA((i + 1) << 5, cur ^ 1);
            stageB((i + 1) << 5, cur ^ 1);
#if USE_TDM
            if (wid == 0) __builtin_amdgcn_s_wait_tensorcnt(1);  // tile i resident (in-order)
#endif
        } else {
#if USE_TDM
            if (wid == 0) __builtin_amdgcn_s_wait_tensorcnt(0);
#endif
        }
        __syncthreads();

        // A fragment (16x32 bf16): lanes 0-15 M=lane, K{0-7,16-23}; lanes 16-31 K{8-15,24-31}
        auto ldA = [&](int rowblk) -> v16bf {
            const uint4* pa = (const uint4*)(lA[cur] + (size_t)(rowblk + mn) * 32);
            const uint4 a0 = pa[half], a1 = pa[2 + half];
            B32x8 w;
            w.d[0] = a0.x; w.d[1] = a0.y; w.d[2] = a0.z; w.d[3] = a0.w;
            w.d[4] = a1.x; w.d[5] = a1.y; w.d[6] = a1.z; w.d[7] = a1.w;
            return __builtin_bit_cast(v16bf, w);
        };
        // B fragment (32x16 bf16): lanes 0-15 N=lane, K{0-7,16-23}; lanes 16-31 K{8-15,24-31}
        auto ldB = [&](int colblk) -> v16bf {
            const uint4* pb = (const uint4*)(lB[cur] + (size_t)(colblk + mn) * 32);
            const uint4 b0 = pb[half], b1 = pb[2 + half];
            B32x8 w;
            w.d[0] = b0.x; w.d[1] = b0.y; w.d[2] = b0.z; w.d[3] = b0.w;
            w.d[4] = b1.x; w.d[5] = b1.y; w.d[6] = b1.z; w.d[7] = b1.w;
            return __builtin_bit_cast(v16bf, w);
        };
        const v16bf af0 = ldA(wm), af1 = ldA(wm + 16);
        const v16bf bf0 = ldB(wn), bf1 = ldB(wn + 16);
        c00 = __builtin_amdgcn_wmma_f32_16x16x32_bf16(false, af0, false, bf0, (short)0, c00, false, false);
        c01 = __builtin_amdgcn_wmma_f32_16x16x32_bf16(false, af0, false, bf1, (short)0, c01, false, false);
        c10 = __builtin_amdgcn_wmma_f32_16x16x32_bf16(false, af1, false, bf0, (short)0, c10, false, false);
        c11 = __builtin_amdgcn_wmma_f32_16x16x32_bf16(false, af1, false, bf1, (short)0, c11, false, false);
        __syncthreads();
    }

    // --- epilogue: C/D layout: VGPR r -> M = r + half*8; N = mn ---
    auto emit = [&](const v8f& acc, int rbase, int cbase) {
        #pragma unroll
        for (int r = 0; r < 8; ++r) {
            const int row = rbase + r;
            const int col = cbase;
            float v = acc[r] + bias[col];
            if (EPI == 1) v += node_emb[node_types[row] * DH + col];
            C[(size_t)row * Nn + col] = v;
        }
    };
    const int r0 = bm + wm + half * 8;
    const int cA = bn + wn + mn, cB = bn + wn + 16 + mn;
    emit(c00, r0,      cA);
    emit(c01, r0,      cB);
    emit(c10, r0 + 16, cA);
    emit(c11, r0 + 16, cB);
}

// ---------------- edge logits: one block per edge, one wave per head ----------------
__global__ __launch_bounds__(256)
void k_logits(const float* __restrict__ xl, const float* __restrict__ xr,
              const float* __restrict__ ee_type, const float* __restrict__ typecnt,
              const float* __restrict__ cnt,
              const int* __restrict__ src_a, const int* __restrict__ dst_a,
              const int* __restrict__ attr, const float* __restrict__ att_l,
              float* __restrict__ logits, u32* __restrict__ lmax) {
    const int e    = blockIdx.x;
    const int head = threadIdx.x >> 5;
    const int lane = threadIdx.x & 31;

    int s, d;
    const bool isLoop = (e >= EE);
    const float* ee = nullptr;
    float tc0=0, tc1=0, tc2=0, tc3=0, tc4=0, ic=1.f;
    if (isLoop) {
        s = d = e - EE;
        tc0 = typecnt[s*5+0]; tc1 = typecnt[s*5+1]; tc2 = typecnt[s*5+2];
        tc3 = typecnt[s*5+3]; tc4 = typecnt[s*5+4];
        ic = 1.0f / fmaxf(cnt[s], 1.0f);
    } else {
        s = src_a[e]; d = dst_a[e];
        ee = ee_type + (size_t)attr[e] * HD;
    }

    float acc = 0.f;
    #pragma unroll
    for (int j = 0; j < 8; ++j) {
        const int cl = lane + j * 32;          // channel within head
        const int hc = head * DH + cl;
        float ev;
        if (isLoop) {
            ev = (tc0 * ee_type[0*HD + hc] + tc1 * ee_type[1*HD + hc] +
                  tc2 * ee_type[2*HD + hc] + tc3 * ee_type[3*HD + hc] +
                  tc4 * ee_type[4*HD + hc]) * ic;
        } else {
            ev = ee[hc];
        }
        float z = xl[(size_t)s * HD + hc] + xr[(size_t)d * HD + hc] + ev;
        z = (z > 0.f) ? z : NEG_SLOPE * z;     // leaky_relu
        acc = fmaf(z, att_l[head * DH + cl], acc);
    }
    // wave32 reduction
    #pragma unroll
    for (int off = 16; off > 0; off >>= 1) acc += __shfl_down(acc, off, 32);
    if (lane == 0) {
        logits[(size_t)e * NH + head] = acc;
        atomicMax(&lmax[d * NH + head], enc_f32(acc));
    }
}

// exp(logit - segmax) ; segment-sum of numerators
__global__ void k_pnum(const float* __restrict__ logits, const u32* __restrict__ lmax,
                       const int* __restrict__ dst_a,
                       float* __restrict__ pnum, float* __restrict__ denom) {
    int i = blockIdx.x * blockDim.x + threadIdx.x;
    if (i >= E2 * NH) return;
    int e = i / NH, h = i % NH;
    int d = (e >= EE) ? (e - EE) : dst_a[e];
    float m = dec_f32(lmax[d * NH + h]);
    float p = __expf(logits[i] - m);
    pnum[i] = p;
    atomicAdd(&denom[d * NH + h], p);
}

// scatter-add of alpha-weighted xl[src], folding the head sum (mean applied at finalize)
__global__ __launch_bounds__(256)
void k_aggregate(const float* __restrict__ xl, const float* __restrict__ pnum,
                 const float* __restrict__ denom,
                 const int* __restrict__ src_a, const int* __restrict__ dst_a,
                 float* __restrict__ acc) {
    const int e = blockIdx.x;
    const int c = threadIdx.x;    // 0..255
    int s, d;
    if (e >= EE) { s = d = e - EE; } else { s = src_a[e]; d = dst_a[e]; }
    float v = 0.f;
    #pragma unroll
    for (int h = 0; h < NH; ++h) {
        float alpha = pnum[(size_t)e * NH + h] / denom[d * NH + h];
        v = fmaf(alpha, xl[(size_t)s * HD + h * DH + c], v);
    }
    atomicAdd(&acc[(size_t)d * DH + c], v);
}

// h = relu(mean_heads + bias); also emit bf16 copy for the next WMMA GEMM
__global__ void k_finalize(const float* __restrict__ acc, const float* __restrict__ b_gat_l,
                           float* __restrict__ h, u16* __restrict__ h_bf) {
    int i = blockIdx.x * blockDim.x + threadIdx.x;
    if (i >= NN * DH) return;
    int c = i % DH;
    float v = fmaxf(acc[i] * (1.0f / NH) + b_gat_l[c], 0.0f);
    h[i] = v;
    h_bf[i] = f32_to_bf16(v);
}

// ---------------- launch ----------------
static inline size_t alignup(size_t x) { return (x + 255) & ~(size_t)255; }

extern "C" void kernel_launch(void* const* d_in, const int* in_sizes, int n_in,
                              void* d_out, int out_size, void* d_ws, size_t ws_size,
                              hipStream_t stream) {
    (void)in_sizes; (void)n_in; (void)out_size; (void)ws_size;
    const float* x         = (const float*)d_in[0];
    const int*   ei        = (const int*)  d_in[1];
    const int*   eattr     = (const int*)  d_in[2];
    const int*   ntypes    = (const int*)  d_in[3];
    const float* W_in      = (const float*)d_in[4];
    const float* b_in      = (const float*)d_in[5];
    const float* node_emb  = (const float*)d_in[6];
    const float* edge_emb  = (const float*)d_in[7];
    const float* Wl        = (const float*)d_in[8];
    const float* bl        = (const float*)d_in[9];
    const float* Wr        = (const float*)d_in[10];
    const float* br        = (const float*)d_in[11];
    const float* We        = (const float*)d_in[12];
    const float* att       = (const float*)d_in[13];
    const float* b_gat     = (const float*)d_in[14];
    const float* W_out     = (const float*)d_in[15];
    const float* b_out     = (const float*)d_in[16];
    const int* src = ei;
    const int* dst = ei + EE;

    // workspace carve-out
    char* w = (char*)d_ws; size_t off = 0;
    const size_t wslot = (size_t)DH * HD * 2;   // 1MB, >= DIN*DH*2
    u16*   x_bf    = (u16*)  (w + off); off = alignup(off + (size_t)NN * DIN * 2);
    float* h       = (float*)(w + off); off = alignup(off + (size_t)NN * DH * 4);
    u16*   h_bf    = (u16*)  (w + off); off = alignup(off + (size_t)NN * DH * 2);
    u16*   wA_bf   = (u16*)  (w + off); off = alignup(off + wslot);
    u16*   wB_bf   = (u16*)  (w + off); off = alignup(off + wslot);
    float* xl      = (float*)(w + off); off = alignup(off + (size_t)NN * HD * 4);
    float* xr      = (float*)(w + off); off = alignup(off + (size_t)NN * HD * 4);
    float* ee_type = (float*)(w + off); off = alignup(off + (size_t)5 * HD * 4);
    float* typecnt = (float*)(w + off); off = alignup(off + (size_t)NN * 5 * 4);
    float* cnt     = (float*)(w + off); off = alignup(off + (size_t)NN * 4);
    float* logits  = (float*)(w + off); off = alignup(off + (size_t)E2 * NH * 4);
    float* pnum    = (float*)(w + off); off = alignup(off + (size_t)E2 * NH * 4);
    u32*   lmax    = (u32*)  (w + off); off = alignup(off + (size_t)NN * NH * 4);
    float* denom   = (float*)(w + off); off = alignup(off + (size_t)NN * NH * 4);
    float* acc     = (float*)(w + off); off = alignup(off + (size_t)NN * DH * 4);

    const int T = 256;
    auto blks = [](long n) { return (int)((n + 255) / 256); };

    // precompute: bf16 inputs + incoming-edge type histogram
    k_cvt_bf16<<<blks((long)NN * DIN), T, 0, stream>>>(x, x_bf, NN * DIN);
    k_cvt_bf16<<<blks((long)DIN * DH), T, 0, stream>>>(W_in, wA_bf, DIN * DH);
    k_fill_f32<<<blks((long)NN * 5), T, 0, stream>>>(typecnt, 0.f, NN * 5);
    k_fill_f32<<<blks(NN), T, 0, stream>>>(cnt, 0.f, NN);
    k_typecnt<<<blks(EE), T, 0, stream>>>(dst, eattr, typecnt, cnt);

    // h0 = x @ W_in + b_in + node_emb[node_types]
    k_wmma_gemm<1><<<dim3(DH / 64, NN / 128), T, 0, stream>>>(
        x_bf, wA_bf, b_in, h, NN, DIN, DH, ntypes, node_emb);
    k_cvt_bf16<<<blks((long)NN * DH), T, 0, stream>>>(h, h_bf, NN * DH);

    for (int l = 0; l < NL; ++l) {
        const float* Wl_l = Wl + (size_t)l * DH * HD;
        const float* Wr_l = Wr + (size_t)l * DH * HD;
        const float* We_l = We + (size_t)l * DH * HD;
        const float* bl_l = bl + (size_t)l * HD;
        const float* br_l = br + (size_t)l * HD;
        const float* att_l = att + (size_t)l * NH * DH;
        const float* bg_l  = b_gat + (size_t)l * DH;

        k_cvt_bf16<<<blks((long)DH * HD), T, 0, stream>>>(Wl_l, wA_bf, DH * HD);
        k_cvt_bf16<<<blks((long)DH * HD), T, 0, stream>>>(Wr_l, wB_bf, DH * HD);
        k_edge_type_proj<<<blks((long)5 * HD), T, 0, stream>>>(edge_emb, We_l, ee_type);

        // xl = h@Wl + bl ; xr = h@Wr + br   (8192 x 256 x 2048 bf16 WMMA)
        k_wmma_gemm<0><<<dim3(HD / 64, NN / 128), T, 0, stream>>>(
            h_bf, wA_bf, bl_l, xl, NN, DH, HD, nullptr, nullptr);
        k_wmma_gemm<0><<<dim3(HD / 64, NN / 128), T, 0, stream>>>(
            h_bf, wB_bf, br_l, xr, NN, DH, HD, nullptr, nullptr);

        k_fill_f32<<<blks((long)NN * NH), T, 0, stream>>>((float*)lmax, 0.f, NN * NH);
        k_fill_f32<<<blks((long)NN * NH), T, 0, stream>>>(denom, 0.f, NN * NH);
        k_fill_f32<<<blks((long)NN * DH), T, 0, stream>>>(acc, 0.f, NN * DH);

        k_logits<<<E2, T, 0, stream>>>(xl, xr, ee_type, typecnt, cnt,
                                       src, dst, eattr, att_l, logits, lmax);
        k_pnum<<<blks((long)E2 * NH), T, 0, stream>>>(logits, lmax, dst, pnum, denom);
        k_aggregate<<<E2, T, 0, stream>>>(xl, pnum, denom, src, dst, acc);
        k_finalize<<<blks((long)NN * DH), T, 0, stream>>>(acc, bg_l, h, h_bf);
    }

    // out = h @ W_out + b_out
    k_cvt_bf16<<<blks((long)DH * DH), T, 0, stream>>>(W_out, wA_bf, DH * DH);
    k_wmma_gemm<0><<<dim3(DH / 64, NN / 128), T, 0, stream>>>(
        h_bf, wA_bf, b_out, (float*)d_out, NN, DH, DH, nullptr, nullptr);
}
